// NeuralEpisodicControl_24601572671491
// MI455X (gfx1250) — matrix operand order repaired and checked by reading, not compiled, not run on hardware
//
#include <hip/hip_runtime.h>
#include <hip/hip_bf16.h>

typedef _Float16 half_t;
typedef __attribute__((ext_vector_type(16))) _Float16 v16h;
typedef __attribute__((ext_vector_type(8)))  _Float16 v8h;
typedef __attribute__((ext_vector_type(8)))  float    v8f;

#define WMMA_F16(a, b, c) \
  __builtin_amdgcn_wmma_f32_16x16x32_f16(false, (a), false, (b), (short)0, (c), false, false)

#define KNN_K 50
#define KNN_ROWS 32   // q-rows per workgroup in the kNN kernel

// ---------------------------------------------------------------------------
// f32 -> f16 elementwise conversion
// ---------------------------------------------------------------------------
__global__ void cvt_f32_f16(const float* __restrict__ src, half_t* __restrict__ dst, int n) {
  int i = blockIdx.x * blockDim.x + threadIdx.x;
  int stride = gridDim.x * blockDim.x;
  for (; i < n; i += stride) dst[i] = (half_t)src[i];
}

// ---------------------------------------------------------------------------
// mem_keys: transpose to keysT[D][M] (f16) + per-row sum of squares (f32).
// One wave per key row; D=128 -> 4 elements/lane.
// ---------------------------------------------------------------------------
__global__ void prep_keys(const float* __restrict__ keys, half_t* __restrict__ keysT,
                          float* __restrict__ k2, int M, int D) {
  int gw = (int)((blockIdx.x * blockDim.x + threadIdx.x) >> 5);
  int lane = threadIdx.x & 31;
  if (gw >= M) return;
  const float* row = keys + (size_t)gw * D;
  float ss = 0.f;
  for (int i = lane; i < D; i += 32) {
    float v = row[i];
    ss += v * v;
    keysT[(size_t)i * M + gw] = (half_t)v;
  }
#pragma unroll
  for (int off = 16; off > 0; off >>= 1) ss += __shfl_down(ss, off, 32);
  if (lane == 0) k2[gw] = ss;
}

// ---------------------------------------------------------------------------
// Generic WMMA GEMM: out = act(A[M,K] @ W[K,N] + bias), f16 inputs, f32 acc.
// Block = 256 threads = 8 waves; wave w owns N-tile blockIdx.y*128 + w*16;
// blockIdx.x owns a 16-row M tile. K must be a multiple of 32.
// Fragment layouts per CDNA5 ISA 7.12.2:
//   A (16x32 f16): lane l -> row M = l%16, K chunks (l/16)*8 + {0..7, 16..23}
//   B (32x16 f16): lane l -> col N = l%16, K chunks (l/16)*8 + {0..7, 16..23} (strided loads from W)
//   C/D (16x16 f32): VGPR r -> M = r + 8*(l/16), N = l%16
// ---------------------------------------------------------------------------
__global__ __launch_bounds__(256) void gemm_wmma_f16(
    const half_t* __restrict__ A, const half_t* __restrict__ W,
    const float* __restrict__ bias, half_t* __restrict__ out16,
    float* __restrict__ out32, int M, int K, int N, int do_relu) {
  int wave = threadIdx.x >> 5, lane = threadIdx.x & 31;
  int row0 = blockIdx.x << 4;
  int col0 = (blockIdx.y << 7) + (wave << 4);
  if (col0 >= N) return;

  int arow = row0 + (lane & 15);
  int koff = (lane >> 4) << 3;
  const half_t* arp = A + (size_t)arow * K + koff;

  v8f c = {};
  for (int kt = 0; kt < K; kt += 32) {
    const half_t* ap = arp + kt;
    v8h lo = *(const v8h*)ap;
    v8h hi = *(const v8h*)(ap + 16);
    v16h a;
#pragma unroll
    for (int i = 0; i < 8; i++) { a[i] = lo[i]; a[i + 8] = hi[i]; }
    v16h b;
#pragma unroll
    for (int i = 0; i < 8; i++) {
      b[i] = W[(size_t)(kt + koff + i) * N + col0 + (lane & 15)];
      b[i + 8] = W[(size_t)(kt + koff + 16 + i) * N + col0 + (lane & 15)];
    }
    c = WMMA_F16(a, b, c);
  }

  int col = col0 + (lane & 15);
  float bv = bias ? bias[col] : 0.f;
  int rbase = row0 + ((lane >> 4) << 3);
#pragma unroll
  for (int r = 0; r < 8; r++) {
    float v = c[r] + bv;
    if (do_relu) v = v > 0.f ? v : 0.f;
    size_t o = (size_t)(rbase + r) * N + col;
    if (out32) out32[o] = v;
    if (out16) out16[o] = (half_t)v;
  }
}

// ---------------------------------------------------------------------------
// LayerNorm over D=128 per row (one wave per row) + q2[row] = sum(q*q).
// Emits q in f16 for the WMMA distance GEMM.
// ---------------------------------------------------------------------------
__global__ void ln_kernel(const float* __restrict__ x, const float* __restrict__ g,
                          const float* __restrict__ b, half_t* __restrict__ q16,
                          float* __restrict__ q2, int B) {
  int row = (int)((blockIdx.x * blockDim.x + threadIdx.x) >> 5);
  int lane = threadIdx.x & 31;
  if (row >= B) return;
  const float* xr = x + (size_t)row * 128;
  float v[4], s = 0.f, s2 = 0.f;
#pragma unroll
  for (int j = 0; j < 4; j++) { v[j] = xr[j * 32 + lane]; s += v[j]; s2 += v[j] * v[j]; }
#pragma unroll
  for (int off = 16; off > 0; off >>= 1) { s += __shfl_down(s, off, 32); s2 += __shfl_down(s2, off, 32); }
  s = __shfl(s, 0, 32);
  s2 = __shfl(s2, 0, 32);
  float mean = s * (1.f / 128.f);
  float var = s2 * (1.f / 128.f) - mean * mean;
  float rstd = rsqrtf(var + 1e-5f);
  float yy = 0.f;
#pragma unroll
  for (int j = 0; j < 4; j++) {
    int i = j * 32 + lane;
    float y = (v[j] - mean) * rstd * g[i] + b[i];
    q16[(size_t)row * 128 + i] = (half_t)y;
    yy += y * y;
  }
#pragma unroll
  for (int off = 16; off > 0; off >>= 1) yy += __shfl_down(yy, off, 32);
  if (lane == 0) q2[row] = yy;
}

// ---------------------------------------------------------------------------
// net_val[b] = dot(v2[b,:], V3) + c3   (D=128, one wave per row)
// ---------------------------------------------------------------------------
__global__ void gemv_v3(const half_t* __restrict__ v2, const float* __restrict__ V3,
                        const float* __restrict__ c3, float* __restrict__ netv, int B) {
  int row = (int)((blockIdx.x * blockDim.x + threadIdx.x) >> 5);
  int lane = threadIdx.x & 31;
  if (row >= B) return;
  float s = 0.f;
#pragma unroll
  for (int j = 0; j < 4; j++) {
    int i = j * 32 + lane;
    s += (float)v2[(size_t)row * 128 + i] * V3[i];
  }
#pragma unroll
  for (int off = 16; off > 0; off >>= 1) s += __shfl_down(s, off, 32);
  if (lane == 0) netv[row] = s + c3[0];
}

// ---------------------------------------------------------------------------
// Fused distance-GEMM + streaming top-50 + inverse-distance weighting + blend.
// Block = 8 waves = 32 q-rows (two 16-row WMMA M-tiles). Per iteration each
// wave loads one B-fragment per K-tile for its 16 keys and feeds it to TWO
// chained WMMAs (one per M-tile), doubling arithmetic intensity per key byte:
// L2 key traffic = (B/32) blocks * 12.8 MB = 1.6 GB instead of 3.3 GB.
// Distances land in LDS; each wave then owns 4 rows and performs
// threshold-guarded insertion into a per-row 50-entry replace-max buffer
// (expected inserts ~ 50*ln(M/50) ~ 345/row, so selection stays cheap).
// ---------------------------------------------------------------------------
__global__ __launch_bounds__(256) void knn_kernel(
    const half_t* __restrict__ q16, const float* __restrict__ q2,
    const half_t* __restrict__ keysT, const float* __restrict__ k2,
    const float* __restrict__ memv, const float* __restrict__ netv,
    float* __restrict__ out, int M) {
  __shared__ float s_dist[KNN_ROWS][132];
  __shared__ float s_hd[KNN_ROWS][64];
  __shared__ int   s_hi[KNN_ROWS][64];
  __shared__ float s_th[KNN_ROWS];
  __shared__ int   s_mp[KNN_ROWS];

  int wave = threadIdx.x >> 5, lane = threadIdx.x & 31;
  int row0 = blockIdx.x * KNN_ROWS;

  for (int i = threadIdx.x; i < KNN_ROWS * 64; i += 256) {
    (&s_hd[0][0])[i] = 3.0e38f;
    (&s_hi[0][0])[i] = 0;
  }
  if (threadIdx.x < KNN_ROWS) { s_th[threadIdx.x] = 3.0e38f; s_mp[threadIdx.x] = 0; }
  __syncthreads();

  // Preload q A-fragments: 2 M-tiles x 4 K-tiles (K=128).
  int koff = (lane >> 4) << 3;
  int rbase = (lane >> 4) << 3;
  v16h afr[2][4];
  float q2v[2][8];
#pragma unroll
  for (int mt = 0; mt < 2; mt++) {
    int arow = row0 + mt * 16 + (lane & 15);
    const half_t* qp = q16 + (size_t)arow * 128 + koff;
#pragma unroll
    for (int kt = 0; kt < 4; kt++) {
      v8h lo = *(const v8h*)(qp + kt * 32);
      v8h hi = *(const v8h*)(qp + kt * 32 + 16);
#pragma unroll
      for (int i = 0; i < 8; i++) { afr[mt][kt][i] = lo[i]; afr[mt][kt][i + 8] = hi[i]; }
    }
#pragma unroll
    for (int r = 0; r < 8; r++) q2v[mt][r] = q2[row0 + mt * 16 + rbase + r];
  }

  int ntile = (M + 15) >> 4;     // 16-key tiles
  int niter = (ntile + 7) >> 3;  // 8 tiles (128 keys) per block iteration

  for (int it = 0; it < niter; ++it) {
    int keybase = ((it << 3) + wave) << 4;
    if (keybase < M) {
      int nkb = keybase + 128;
      if (nkb < M) __builtin_prefetch(keysT + (size_t)lane * M + nkb, 0, 0);
      v8f c0 = {};
      v8f c1 = {};
#pragma unroll
      for (int kt = 0; kt < 4; kt++) {
        v16h b;
#pragma unroll
        for (int i = 0; i < 8; i++) {
          b[i] = keysT[(size_t)((kt << 5) + koff + i) * M + keybase + (lane & 15)];
          b[i + 8] = keysT[(size_t)((kt << 5) + koff + 16 + i) * M + keybase + (lane & 15)];
        }
        c0 = WMMA_F16(afr[0][kt], b, c0);
        c1 = WMMA_F16(afr[1][kt], b, c1);
      }
      float k2c = k2[keybase + (lane & 15)];
      int colw = (wave << 4) + (lane & 15);
#pragma unroll
      for (int r = 0; r < 8; r++) {
        s_dist[rbase + r][colw]      = q2v[0][r] + k2c - 2.0f * c0[r];
        s_dist[16 + rbase + r][colw] = q2v[1][r] + k2c - 2.0f * c1[r];
      }
    }
    __syncthreads();

    int ncand = M - (it << 7);
    if (ncand > 128) ncand = 128;
    for (int rsel = 0; rsel < 4; rsel++) {
      int rr = (wave << 2) + rsel;
      float th = s_th[rr];
#pragma unroll
      for (int j = 0; j < 4; j++) {
        int cidx = (j << 5) + lane;
        float d = (cidx < ncand) ? s_dist[rr][cidx] : 3.0e38f;
        unsigned long long mask = __ballot(d < th);
        while (mask) {
          int src = __builtin_ctzll(mask);
          mask &= mask - 1;
          float dc = __shfl(d, src, 32);
          if (!(dc < th)) continue;  // threshold may have tightened
          int kc = (it << 7) + __shfl(cidx, src, 32);
          int mp = s_mp[rr];
          if (lane == 0) { s_hd[rr][mp] = dc; s_hi[rr][mp] = kc; }
          // Rescan 50 slots for new max (slot mp substituted locally with dc).
          float m0 = (lane < KNN_K) ? ((lane == mp) ? dc : s_hd[rr][lane]) : -3.0e38f;
          int i0 = lane;
          int l2 = lane + 32;
          float m1 = (l2 < KNN_K) ? ((l2 == mp) ? dc : s_hd[rr][l2]) : -3.0e38f;
          if (m1 > m0) { m0 = m1; i0 = l2; }
#pragma unroll
          for (int off = 16; off > 0; off >>= 1) {
            float om = __shfl_down(m0, off, 32);
            int oi = __shfl_down(i0, off, 32);
            if (om > m0) { m0 = om; i0 = oi; }
          }
          m0 = __shfl(m0, 0, 32);
          i0 = __shfl(i0, 0, 32);
          if (lane == 0) { s_th[rr] = m0; s_mp[rr] = i0; }
          th = m0;
        }
      }
    }
    __syncthreads();
  }

  // Finalize: inverse-distance weighting over the 50 kept entries + blend.
  for (int rsel = 0; rsel < 4; rsel++) {
    int rr = (wave << 2) + rsel;
    float wsum = 0.f, vsum = 0.f;
    for (int s = lane; s < KNN_K; s += 32) {
      float dd = s_hd[rr][s];
      float ww = 1.0f / (dd + 1e-7f);
      wsum += ww;
      vsum += ww * memv[s_hi[rr][s]];
    }
#pragma unroll
    for (int off = 16; off > 0; off >>= 1) {
      wsum += __shfl_down(wsum, off, 32);
      vsum += __shfl_down(vsum, off, 32);
    }
    if (lane == 0) {
      int grow = row0 + rr;
      out[grow] = 0.9f * (vsum / wsum) + 0.1f * netv[grow];
    }
  }
}

// ---------------------------------------------------------------------------
// Host launcher
// ---------------------------------------------------------------------------
extern "C" void kernel_launch(void* const* d_in, const int* in_sizes, int n_in,
                              void* d_out, int out_size, void* d_ws, size_t ws_size,
                              hipStream_t stream) {
  (void)in_sizes; (void)n_in; (void)out_size; (void)ws_size;
  const float* states     = (const float*)d_in[0];
  const float* W1         = (const float*)d_in[1];
  const float* b1         = (const float*)d_in[2];
  const float* W2         = (const float*)d_in[3];
  const float* b2         = (const float*)d_in[4];
  const float* ln_g       = (const float*)d_in[5];
  const float* ln_b       = (const float*)d_in[6];
  const float* mem_keys   = (const float*)d_in[7];
  const float* mem_values = (const float*)d_in[8];
  const float* V1         = (const float*)d_in[9];
  const float* c1         = (const float*)d_in[10];
  const float* V2         = (const float*)d_in[11];
  const float* c2         = (const float*)d_in[12];
  const float* V3         = (const float*)d_in[13];
  const float* c3         = (const float*)d_in[14];
  float* out = (float*)d_out;

  const int B = 4096, S = 544, H = 256, D = 128, M = 50000;

  char* p = (char*)d_ws;
  auto carve = [&](size_t bytes) -> void* {
    void* r = (void*)p;
    p += (bytes + 255) & ~(size_t)255;
    return r;
  };
  half_t* states16 = (half_t*)carve((size_t)B * S * 2);
  half_t* W1_16    = (half_t*)carve((size_t)S * H * 2);
  half_t* V1_16    = (half_t*)carve((size_t)S * H * 2);
  half_t* W2_16    = (half_t*)carve((size_t)H * D * 2);
  half_t* V2_16    = (half_t*)carve((size_t)H * D * 2);
  half_t* h16      = (half_t*)carve((size_t)B * H * 2);
  half_t* v1_16    = (half_t*)carve((size_t)B * H * 2);
  float*  qpre     = (float*)carve((size_t)B * D * 4);
  half_t* q16      = (half_t*)carve((size_t)B * D * 2);
  float*  q2       = (float*)carve((size_t)B * 4);
  half_t* v2_16    = (half_t*)carve((size_t)B * D * 2);
  float*  netv     = (float*)carve((size_t)B * 4);
  half_t* keysT    = (half_t*)carve((size_t)D * M * 2);
  float*  k2       = (float*)carve((size_t)M * 4);

  cvt_f32_f16<<<1024, 256, 0, stream>>>(states, states16, B * S);
  cvt_f32_f16<<<256, 256, 0, stream>>>(W1, W1_16, S * H);
  cvt_f32_f16<<<256, 256, 0, stream>>>(V1, V1_16, S * H);
  cvt_f32_f16<<<128, 256, 0, stream>>>(W2, W2_16, H * D);
  cvt_f32_f16<<<128, 256, 0, stream>>>(V2, V2_16, H * D);
  prep_keys<<<(M + 7) / 8, 256, 0, stream>>>(mem_keys, keysT, k2, M, D);

  dim3 g1(B / 16, H / 128);  // encoder / value layer 1: N=256
  gemm_wmma_f16<<<g1, 256, 0, stream>>>(states16, W1_16, b1, h16, nullptr, B, S, H, 1);
  gemm_wmma_f16<<<g1, 256, 0, stream>>>(states16, V1_16, c1, v1_16, nullptr, B, S, H, 1);

  dim3 g2(B / 16, 1);        // layer 2: N=128
  gemm_wmma_f16<<<g2, 256, 0, stream>>>(h16, W2_16, b2, nullptr, qpre, B, H, D, 0);
  gemm_wmma_f16<<<g2, 256, 0, stream>>>(v1_16, V2_16, c2, v2_16, nullptr, B, H, D, 1);

  ln_kernel<<<B / 8, 256, 0, stream>>>(qpre, ln_g, ln_b, q16, q2, B);
  gemv_v3<<<B / 8, 256, 0, stream>>>(v2_16, V3, c3, netv, B);

  knn_kernel<<<B / KNN_ROWS, 256, 0, stream>>>(q16, q2, keysT, k2, mem_values, netv, out, M);
}
